// dynamic_regional_graph_62612033241632
// MI455X (gfx1250) — compile-verified, matrix operand order checked by reading, not run on hardware
//
#include <hip/hip_runtime.h>
#include <hip/hip_bf16.h>
#include <math.h>

typedef __attribute__((ext_vector_type(16))) _Float16 v16h;
typedef __attribute__((ext_vector_type(8)))  _Float16 v8h;
typedef __attribute__((ext_vector_type(4)))  _Float16 v4h;
typedef __attribute__((ext_vector_type(8)))  float    v8f;
typedef __attribute__((ext_vector_type(4)))  float    v4f;

#define BB   4
#define SS   4096
#define DD   128
#define NSPK 9
#define NT   (SS / 16)      // 256 row tiles
#define WIN  15

// ---------------------------------------------------------------------------
// Load one 16x32 f16 A/B fragment per CDNA5 ISA layout:
//   lanes 0-15 : row = base+lane, K in {0..7, 16..23}
//   lanes16-31 : row = base+lane-16, K in {8..15, 24..31}
// Two contiguous 16B loads per lane.
// ---------------------------------------------------------------------------
__device__ inline v16h load_frag(const _Float16* __restrict__ rowbase,
                                 int kb, int lane) {
    const int hi = (lane >= 16) ? 8 : 0;
    const v8h lo  = *reinterpret_cast<const v8h*>(rowbase + kb * 32 + hi);
    const v8h hiv = *reinterpret_cast<const v8h*>(rowbase + kb * 32 + 16 + hi);
    v16h f;
#pragma unroll
    for (int e = 0; e < 8; ++e) { f[e] = lo[e]; f[8 + e] = hiv[e]; }
    return f;
}

// ---------------------------------------------------------------------------
// Kernel 1: row-normalize x into f16 (xn = x / max(||x||, 1e-8)).
// One wave (32 lanes) per row of 128 floats; 8 rows per 256-thread block.
// ---------------------------------------------------------------------------
__global__ __launch_bounds__(256) void norm_kernel(const float* __restrict__ x,
                                                   _Float16* __restrict__ xn) {
    const int row  = blockIdx.x * 8 + (threadIdx.x >> 5);   // 0 .. B*S-1
    const int lane = threadIdx.x & 31;
    const v4f v = *reinterpret_cast<const v4f*>(x + (size_t)row * DD + lane * 4);
    float ss = v[0] * v[0] + v[1] * v[1] + v[2] * v[2] + v[3] * v[3];
#pragma unroll
    for (int m = 1; m < 32; m <<= 1) ss += __shfl_xor(ss, m, 32);
    const float rn = 1.0f / fmaxf(sqrtf(ss), 1e-8f);
    v4h o;
#pragma unroll
    for (int e = 0; e < 4; ++e) o[e] = (_Float16)(v[e] * rn);
    *reinterpret_cast<v4h*>(xn + (size_t)row * DD + lane * 4) = o;
}

// ---------------------------------------------------------------------------
// Kernel 2: speakers[b][s] = argmax_k qmask[s][b][k]  (first max wins)
// ---------------------------------------------------------------------------
__global__ __launch_bounds__(256) void spk_kernel(const float* __restrict__ qmask,
                                                  int* __restrict__ spk) {
    const int g = blockIdx.x * 256 + threadIdx.x;           // b*S + s
    if (g >= BB * SS) return;
    const int b = g / SS, s = g - b * SS;
    const float* q = qmask + ((size_t)s * BB + b) * NSPK;
    int best = 0; float bv = q[0];
#pragma unroll
    for (int k = 1; k < NSPK; ++k) {
        const float v = q[k];
        if (v > bv) { bv = v; best = k; }
    }
    spk[g] = best;
}

// ---------------------------------------------------------------------------
// Kernel 3: banded similarity via WMMA. One wave per (b, row-tile t).
// Computes 16x48 band block (col tiles t-1,t,t+1), same-speaker doubling,
// row degrees -> dinv, and stores pre-normalized band values.
// ---------------------------------------------------------------------------
__global__ __launch_bounds__(32) void band_kernel(const _Float16* __restrict__ xn,
                                                  const int* __restrict__ spk,
                                                  const int* __restrict__ dia_len,
                                                  float* __restrict__ band,
                                                  float* __restrict__ dinv) {
    const int b    = blockIdx.x / NT;
    const int t    = blockIdx.x % NT;
    const int lane = threadIdx.x;
    const int col  = lane & 15;
    const int hi8  = (lane >> 4) * 8;
    const int i0   = t * 16;
    const int len  = dia_len[b];

    const _Float16* xb = xn + (size_t)b * SS * DD;

    // A fragments for this row tile (lane's row = i0 + col)
    const _Float16* arow = xb + (size_t)(i0 + col) * DD;
    v16h a[4];
#pragma unroll
    for (int kb = 0; kb < 4; ++kb) a[kb] = load_frag(arow, kb, lane);

    int  ivals[8], spk_i[8];
    bool ival[8];
#pragma unroll
    for (int r = 0; r < 8; ++r) {
        const int i = i0 + r + hi8;
        ivals[r] = i;
        spk_i[r] = spk[b * SS + i];
        ival[r]  = i < len;
    }

    float    wM[3][8];
    unsigned bits[3];
    int      cntLoc[8] = {0, 0, 0, 0, 0, 0, 0, 0};

#pragma unroll
    for (int s = 0; s < 3; ++s) {
        const int jt = t - 1 + s;
        bits[s] = 0u;
        if (jt < 0 || jt >= NT) {
#pragma unroll
            for (int r = 0; r < 8; ++r) wM[s][r] = 0.0f;
            continue;
        }
        const int j = jt * 16 + col;
        const _Float16* brow = xb + (size_t)j * DD;
        v8f c = {};
#pragma unroll
        for (int kb = 0; kb < 4; ++kb) {
            const v16h bf = load_frag(brow, kb, lane);
            c = __builtin_amdgcn_wmma_f32_16x16x32_f16(
                    false, a[kb], false, bf, (short)0, c, false, false);
        }
        const int  spkj = spk[b * SS + j];
        const bool jval = j < len;
#pragma unroll
        for (int r = 0; r < 8; ++r) {
            const float cs = fminf(1.0f, fmaxf(-1.0f, c[r]));
            const float w  = 1.0f - acosf(cs) * 0.31830988618379067f; // /pi
            const int   d  = ivals[r] - j;
            const bool  M  = (d <= WIN) && (d >= -WIN) && ival[r] && jval;
            const bool  sm = M && (spk_i[r] == spkj);
            bits[s]  |= sm ? (1u << r) : 0u;
            cntLoc[r] += sm ? 1 : 0;
            wM[s][r] = M ? w : 0.0f;
        }
    }

    // same-speaker counts: reduce over the 16-lane half (columns of the row)
    int cnt[8];
#pragma unroll
    for (int r = 0; r < 8; ++r) {
        int v = cntLoc[r];
#pragma unroll
        for (int m = 1; m < 16; m <<= 1) v += __shfl_xor(v, m, 32);
        cnt[r] = v;
    }

    // adjacency with doubling, row degrees
    float adj[3][8];
    float degLoc[8] = {0, 0, 0, 0, 0, 0, 0, 0};
#pragma unroll
    for (int s = 0; s < 3; ++s)
#pragma unroll
        for (int r = 0; r < 8; ++r) {
            const bool dbl = (cnt[r] > 1) && ((bits[s] >> r) & 1u);
            const float v = wM[s][r] * (dbl ? 2.0f : 1.0f);
            adj[s][r] = v;
            degLoc[r] += v;
        }

    float dinvr[8];
#pragma unroll
    for (int r = 0; r < 8; ++r) {
        float v = degLoc[r];
#pragma unroll
        for (int m = 1; m < 16; m <<= 1) v += __shfl_xor(v, m, 32);
        const float deg = (v == 0.0f) ? 1.0f : v;
        dinvr[r] = 1.0f / sqrtf(deg);
    }

    // store band (48 cols per row) and dinv
#pragma unroll
    for (int s = 0; s < 3; ++s)
#pragma unroll
        for (int r = 0; r < 8; ++r)
            band[((size_t)(b * SS + ivals[r])) * 48 + s * 16 + col] = adj[s][r];

    if (col == 0) {
#pragma unroll
        for (int r = 0; r < 8; ++r) dinv[b * SS + ivals[r]] = dinvr[r];
    }
}

// ---------------------------------------------------------------------------
// Kernel 4: stream the 256 MB output. One block per (b,i) row; nontemporal
// float4 stores; band window gets dinv_i * dinv_j * band value, rest zero.
// ---------------------------------------------------------------------------
__global__ __launch_bounds__(256) void write_kernel(const float* __restrict__ band,
                                                    const float* __restrict__ dinv,
                                                    float* __restrict__ out) {
    const int g = blockIdx.x;                // b*S + i
    const int b = g >> 12;                   // /4096
    const int i = g & (SS - 1);
    const int c0 = (i & ~15) - 16;           // first band column
    const float di = dinv[g];
    const float* __restrict__ bandrow = band + (size_t)g * 48;
    const float* __restrict__ dinvb   = dinv + b * SS;
    float* __restrict__ orow = out + (size_t)g * SS;

#pragma unroll
    for (int it = 0; it < 4; ++it) {
        const int j0 = (threadIdx.x + it * 256) * 4;
        v4f v;
#pragma unroll
        for (int e = 0; e < 4; ++e) {
            const int j = j0 + e;
            const int idx = j - c0;
            float val = 0.0f;
            if (idx >= 0 && idx < 48) val = di * dinvb[j] * bandrow[idx];
            v[e] = val;
        }
        __builtin_nontemporal_store(v, reinterpret_cast<v4f*>(orow + j0));
    }
}

// ---------------------------------------------------------------------------
extern "C" void kernel_launch(void* const* d_in, const int* in_sizes, int n_in,
                              void* d_out, int out_size, void* d_ws, size_t ws_size,
                              hipStream_t stream) {
    const float* x       = (const float*)d_in[0];   // [B,S,D]
    const float* qmask   = (const float*)d_in[1];   // [S,B,NSPK]
    const int*   dia_len = (const int*)d_in[2];     // [B]
    float*       out     = (float*)d_out;           // [B,S,S]

    char* ws = (char*)d_ws;
    _Float16* xn   = (_Float16*)ws;                       // 4 MB
    int*      spk  = (int*)(ws + 4194304);                // 64 KB
    float*    dinv = (float*)(ws + 4259840);              // 64 KB
    float*    band = (float*)(ws + 4325376);              // 3 MB

    norm_kernel <<<(BB * SS) / 8, 256, 0, stream>>>(x, xn);
    spk_kernel  <<<(BB * SS + 255) / 256, 256, 0, stream>>>(qmask, spk);
    band_kernel <<<BB * NT, 32, 0, stream>>>(xn, spk, dia_len, band, dinv);
    write_kernel<<<BB * SS, 256, 0, stream>>>(band, dinv, out);
}
